// SGModule_8959301780007
// MI455X (gfx1250) — matrix operand order repaired and checked by reading, not compiled, not run on hardware
//
#include <hip/hip_runtime.h>

// ---------------------------------------------------------------------------
// SGFormer (2-layer linear attention) for MI455X / gfx1250.
// All GEMMs via v_wmma_f32_16x16x32_bf16 (bf16 in, f32 accumulate).
// Wave tile 32x64 (2x4 WMMA sub-tiles), block tile 128x128 (8 waves).
// ---------------------------------------------------------------------------

typedef __attribute__((ext_vector_type(16))) __bf16 v16bf;
typedef __attribute__((ext_vector_type(8)))  __bf16 v8bf;
typedef __attribute__((ext_vector_type(8)))  float  v8f;

#define TOT 65536   // B_GRAPHS * N_NODES
#define INC 256     // IN_CH
#define HID 512     // HID == HC (1 head)
#define NGR 16      // B_GRAPHS
#define NPG 4096    // N_NODES per graph
#define LN_EPS 1e-5f
#define QK_EPS 1e-6f

// ------------------------------ helpers ------------------------------------

__device__ __forceinline__ float wave_sum(float v) {
#pragma unroll
  for (int off = 16; off > 0; off >>= 1) v += __shfl_xor(v, off, 32);
  return v;
}

union AFrag { v16bf v; v8bf h[2]; };

// Accumulate a 32x64 output tile: D += A[M,K] * W[N,K]^T.
// A row-major [*,K] (bf16), W row-major [N,K] (bf16) == column-major B.
// Fragment layouts used here (wave32, same contiguous k order for A and B):
//   A: lane = m%16 + 16*hi ; elems 0..15 -> K=16*hi..16*hi+15 (contiguous)
//   B: lane = n%16 + 16*hi ; elems 0..15 -> K=16*hi..16*hi+15 (contiguous)
//   D: lane = n%16 + 16*hi ; vgpr r -> m = 8*hi + r
__device__ __forceinline__ void wmma_2x4(
    const __bf16* __restrict__ A, const __bf16* __restrict__ W,
    int K, int lane, int rowBase, int colBase, v8f acc[2][4]) {
  const int m16 = lane & 15;
  const int hi  = lane >> 4;
  const __bf16* arow0 = A + (size_t)(rowBase + m16)      * K;
  const __bf16* arow1 = A + (size_t)(rowBase + 16 + m16) * K;
  const __bf16* wrow[4];
#pragma unroll
  for (int j = 0; j < 4; ++j)
    wrow[j] = W + (size_t)(colBase + 16 * j + m16) * K + 16 * hi;
#pragma unroll 2
  for (int k0 = 0; k0 < K; k0 += 32) {
    AFrag a0, a1;
    a0.h[0] = *(const v8bf*)(arow0 + k0 + 16 * hi);
    a0.h[1] = *(const v8bf*)(arow0 + k0 + 16 * hi + 8);
    a1.h[0] = *(const v8bf*)(arow1 + k0 + 16 * hi);
    a1.h[1] = *(const v8bf*)(arow1 + k0 + 16 * hi + 8);
    v16bf b[4];
#pragma unroll
    for (int j = 0; j < 4; ++j) b[j] = *(const v16bf*)(wrow[j] + k0);
#pragma unroll
    for (int j = 0; j < 4; ++j) {
      acc[0][j] = __builtin_amdgcn_wmma_f32_16x16x32_bf16(
          false, a0.v, false, b[j], (short)0, acc[0][j], false, false);
      acc[1][j] = __builtin_amdgcn_wmma_f32_16x16x32_bf16(
          false, a1.v, false, b[j], (short)0, acc[1][j], false, false);
    }
  }
}

// ------------------------------ kernels ------------------------------------

// C[M,N] = A[M,K] @ W[N,K]^T (+ bias[n]); optional batching via blockIdx.z.
// Block tile 128x128: waves arranged 4(M) x 2(N), each wave 32x64.
__global__ __launch_bounds__(256) void gemm_bias_kernel(
    const __bf16* __restrict__ A, const __bf16* __restrict__ W,
    const float* __restrict__ bias, float* __restrict__ C,
    int N, int K, long sA, long sW, long sC) {
  const int lane = threadIdx.x & 31;
  const int wave = threadIdx.x >> 5;
  const int g = blockIdx.z;
  A += (size_t)g * sA;  W += (size_t)g * sW;  C += (size_t)g * sC;
  const int rowBase = blockIdx.y * 128 + (wave & 3) * 32;
  const int colBase = blockIdx.x * 128 + (wave >> 2) * 64;
  v8f acc[2][4] = {};
  wmma_2x4(A, W, K, lane, rowBase, colBase, acc);
  const int m16 = lane & 15, hi = lane >> 4;
#pragma unroll
  for (int i = 0; i < 2; ++i)
#pragma unroll
    for (int j = 0; j < 4; ++j) {
      const int n = colBase + j * 16 + m16;
      const float bv = bias ? bias[n] : 0.0f;
#pragma unroll
      for (int r = 0; r < 8; ++r) {
        const int m = rowBase + i * 16 + 8 * hi + r;
        C[(size_t)m * N + n] = acc[i][j][r] + bv;
      }
    }
}

// Attention output GEMM with fused SGFormer epilogue:
//   acc = q @ kvsT^T ; num = acc + Npg*v ; a = 0.5*(num/denom + h)
__global__ __launch_bounds__(256) void gemm_attn_kernel(
    const __bf16* __restrict__ Qbf, const __bf16* __restrict__ kvsT,
    const float* __restrict__ V32, const float* __restrict__ H32,
    const float* __restrict__ denom, float* __restrict__ Apre) {
  const int lane = threadIdx.x & 31;
  const int wave = threadIdx.x >> 5;
  const int rowBase = blockIdx.y * 128 + (wave & 3) * 32;
  const int colBase = blockIdx.x * 128 + (wave >> 2) * 64;
  const int g = (blockIdx.y * 128) / NPG;           // 128 | NPG
  const __bf16* W = kvsT + (size_t)g * HID * HID;   // kvsT[d][m], row-major
  v8f acc[2][4] = {};
  wmma_2x4(Qbf, W, HID, lane, rowBase, colBase, acc);
  const int m16 = lane & 15, hi = lane >> 4;
  const float Nf = (float)NPG;
#pragma unroll
  for (int i = 0; i < 2; ++i)
#pragma unroll
    for (int j = 0; j < 4; ++j) {
      const int n = colBase + j * 16 + m16;
#pragma unroll
      for (int r = 0; r < 8; ++r) {
        const int m = rowBase + i * 16 + 8 * hi + r;
        const size_t idx = (size_t)m * HID + n;
        const float num = acc[i][j][r] + Nf * V32[idx];
        Apre[idx] = 0.5f * (num / denom[m] + H32[idx]);
      }
    }
}

// Row-wise LayerNorm + ReLU over HID=512; one wave per row.
// Writes f32 result and a bf16 copy for downstream WMMA GEMMs.
__global__ __launch_bounds__(256) void ln_relu_kernel(
    const float* __restrict__ X, const float* __restrict__ gamma,
    const float* __restrict__ beta, float* __restrict__ Y,
    __bf16* __restrict__ Ybf) {
  const int row  = blockIdx.x * 8 + (threadIdx.x >> 5);
  const int lane = threadIdx.x & 31;
  const float* xr = X + (size_t)row * HID;
  float vals[16];
  float s = 0.0f;
#pragma unroll
  for (int i = 0; i < 16; ++i) { vals[i] = xr[lane + 32 * i]; s += vals[i]; }
  const float mu = wave_sum(s) * (1.0f / HID);
  float vv = 0.0f;
#pragma unroll
  for (int i = 0; i < 16; ++i) { const float d = vals[i] - mu; vv += d * d; }
  const float inv = rsqrtf(wave_sum(vv) * (1.0f / HID) + LN_EPS);
#pragma unroll
  for (int i = 0; i < 16; ++i) {
    const int c = lane + 32 * i;
    float y = (vals[i] - mu) * inv * gamma[c] + beta[c];
    y = fmaxf(y, 0.0f);
    Y[(size_t)row * HID + c]   = y;
    Ybf[(size_t)row * HID + c] = (__bf16)y;
  }
}

// Per-node prep: q,k -> zero->eps, L2-normalize; emit Qbf [TOT,HID],
// kT/vT transposed bf16 [NGR][HID][NPG] for the kvs GEMM. One wave per node.
__global__ __launch_bounds__(256) void prep_attn_kernel(
    float* __restrict__ Q32, const float* __restrict__ K32,
    const float* __restrict__ V32, __bf16* __restrict__ Qbf,
    __bf16* __restrict__ kT, __bf16* __restrict__ vT) {
  const int node = blockIdx.x * 8 + (threadIdx.x >> 5);
  const int lane = threadIdx.x & 31;
  const int g = node / NPG, n = node % NPG;
  const size_t base = (size_t)node * HID;
  const size_t tbase = (size_t)g * HID * NPG + n;
  // q
  float qv[16]; float ss = 0.0f;
#pragma unroll
  for (int i = 0; i < 16; ++i) {
    float q = Q32[base + lane + 32 * i];
    if (q == 0.0f) q = QK_EPS;
    qv[i] = q; ss += q * q;
  }
  float inv = rsqrtf(wave_sum(ss));
#pragma unroll
  for (int i = 0; i < 16; ++i) {
    const float qn = qv[i] * inv;
    Q32[base + lane + 32 * i] = qn;
    Qbf[base + lane + 32 * i] = (__bf16)qn;
  }
  // k (store transposed)
  ss = 0.0f;
#pragma unroll
  for (int i = 0; i < 16; ++i) {
    float k = K32[base + lane + 32 * i];
    if (k == 0.0f) k = QK_EPS;
    qv[i] = k; ss += k * k;
  }
  inv = rsqrtf(wave_sum(ss));
#pragma unroll
  for (int i = 0; i < 16; ++i)
    kT[tbase + (size_t)(lane + 32 * i) * NPG] = (__bf16)(qv[i] * inv);
  // v (store transposed)
#pragma unroll
  for (int i = 0; i < 16; ++i)
    vT[tbase + (size_t)(lane + 32 * i) * NPG] = (__bf16)V32[base + lane + 32 * i];
}

// ks_sum[g][m] = sum_n kT[g][m][n]; one wave per (g,m) row of kT.
__global__ __launch_bounds__(256) void kssum_kernel(
    const __bf16* __restrict__ kT, float* __restrict__ ks) {
  const int row  = blockIdx.x * 8 + (threadIdx.x >> 5);   // 0..NGR*HID
  const int lane = threadIdx.x & 31;
  const __bf16* p = kT + (size_t)row * NPG;
  float s = 0.0f;
  for (int i = lane; i < NPG; i += 32) s += (float)p[i];
  s = wave_sum(s);
  if (lane == 0) ks[row] = s;
}

// denom[node] = dot(q_norm[node,:], ks_sum[g]) + NPG; one wave per node.
__global__ __launch_bounds__(256) void denom_kernel(
    const float* __restrict__ Q32, const float* __restrict__ ks,
    float* __restrict__ denom) {
  const int node = blockIdx.x * 8 + (threadIdx.x >> 5);
  const int lane = threadIdx.x & 31;
  const int g = node / NPG;
  float s = 0.0f;
#pragma unroll
  for (int i = 0; i < 16; ++i) {
    const int c = lane + 32 * i;
    s += Q32[(size_t)node * HID + c] * ks[g * HID + c];
  }
  s = wave_sum(s);
  if (lane == 0) denom[node] = s + (float)NPG;
}

__global__ __launch_bounds__(256) void f2bf_kernel(
    const float* __restrict__ x, __bf16* __restrict__ y, long n) {
  const long i = (long)blockIdx.x * blockDim.x + threadIdx.x;
  if (i < n) y[i] = (__bf16)x[i];
}

// ------------------------------ launch -------------------------------------

extern "C" void kernel_launch(void* const* d_in, const int* in_sizes, int n_in,
                              void* d_out, int out_size, void* d_ws, size_t ws_size,
                              hipStream_t stream) {
  const float* x     = (const float*)d_in[0];
  const float* fc0_w = (const float*)d_in[1];
  const float* fc0_b = (const float*)d_in[2];
  const float* ln0_g = (const float*)d_in[3];
  const float* ln0_b = (const float*)d_in[4];
  const float* qkv_w[2][3] = {
      {(const float*)d_in[5],  (const float*)d_in[6],  (const float*)d_in[7]},
      {(const float*)d_in[10], (const float*)d_in[11], (const float*)d_in[12]}};
  const float* ln_g[2] = {(const float*)d_in[8],  (const float*)d_in[13]};
  const float* ln_b[2] = {(const float*)d_in[9],  (const float*)d_in[14]};
  float* out = (float*)d_out;

  // ---- workspace carve-out ----
  char* w = (char*)d_ws;
  auto carve = [&](size_t bytes) -> void* {
    void* p = (void*)w;
    w += (bytes + 255) & ~(size_t)255;
    return p;
  };
  __bf16* Xbf    = (__bf16*)carve((size_t)TOT * INC * 2);
  __bf16* fc0wb  = (__bf16*)carve((size_t)HID * INC * 2);
  __bf16* wb[2][3];
  for (int l = 0; l < 2; ++l)
    for (int m = 0; m < 3; ++m)
      wb[l][m] = (__bf16*)carve((size_t)HID * HID * 2);
  float*  G      = (float*) carve((size_t)TOT * HID * 4);  // current h (f32)
  __bf16* Hbf    = (__bf16*)carve((size_t)TOT * HID * 2);  // current h (bf16)
  float*  Q32    = (float*) carve((size_t)TOT * HID * 4);
  float*  K32    = (float*) carve((size_t)TOT * HID * 4);  // reused as Apre
  float*  V32    = (float*) carve((size_t)TOT * HID * 4);
  __bf16* Qbf    = (__bf16*)carve((size_t)TOT * HID * 2);
  __bf16* kT     = (__bf16*)carve((size_t)NGR * HID * NPG * 2);
  __bf16* vT     = (__bf16*)carve((size_t)NGR * HID * NPG * 2);
  float*  kvsT32 = (float*) carve((size_t)NGR * HID * HID * 4);
  __bf16* kvsTbf = (__bf16*)carve((size_t)NGR * HID * HID * 2);
  float*  ks     = (float*) carve((size_t)NGR * HID * 4);
  float*  den    = (float*) carve((size_t)TOT * 4);
  (void)in_sizes; (void)n_in; (void)out_size; (void)ws_size;

  const dim3 blk(256);
  auto cvt = [&](const float* src, __bf16* dst, long n) {
    f2bf_kernel<<<(unsigned)((n + 255) / 256), blk, 0, stream>>>(src, dst, n);
  };

  // ---- weight / input downcasts ----
  cvt(x, Xbf, (long)TOT * INC);
  cvt(fc0_w, fc0wb, (long)HID * INC);
  for (int l = 0; l < 2; ++l)
    for (int m = 0; m < 3; ++m)
      cvt(qkv_w[l][m], wb[l][m], (long)HID * HID);

  // ---- fc0 + LN0 + ReLU ----
  {
    dim3 grd(HID / 128, TOT / 128, 1);
    gemm_bias_kernel<<<grd, blk, 0, stream>>>(Xbf, fc0wb, fc0_b, G,
                                              HID, INC, 0, 0, 0);
    ln_relu_kernel<<<TOT / 8, blk, 0, stream>>>(G, ln0_g, ln0_b, G, Hbf);
  }

  // ---- attention layers ----
  for (int l = 0; l < 2; ++l) {
    dim3 gqkv(HID / 128, TOT / 128, 1);
    gemm_bias_kernel<<<gqkv, blk, 0, stream>>>(Hbf, wb[l][0], nullptr, Q32,
                                               HID, HID, 0, 0, 0);
    gemm_bias_kernel<<<gqkv, blk, 0, stream>>>(Hbf, wb[l][1], nullptr, K32,
                                               HID, HID, 0, 0, 0);
    gemm_bias_kernel<<<gqkv, blk, 0, stream>>>(Hbf, wb[l][2], nullptr, V32,
                                               HID, HID, 0, 0, 0);

    prep_attn_kernel<<<TOT / 8, blk, 0, stream>>>(Q32, K32, V32, Qbf, kT, vT);
    kssum_kernel<<<(NGR * HID) / 8, blk, 0, stream>>>(kT, ks);
    denom_kernel<<<TOT / 8, blk, 0, stream>>>(Q32, ks, den);

    // kvsT[g][d][m] = sum_n vT[g][d][n] * kT[g][m][n]   (batched WMMA GEMM)
    {
      dim3 grd(HID / 128, HID / 128, NGR);
      gemm_bias_kernel<<<grd, blk, 0, stream>>>(
          vT, kT, nullptr, kvsT32, HID, NPG,
          (long)HID * NPG, (long)HID * NPG, (long)HID * HID);
    }
    cvt(kvsT32, kvsTbf, (long)NGR * HID * HID);

    // num/denom + residual -> Apre (reuse K32)
    {
      dim3 grd(HID / 128, TOT / 128, 1);
      gemm_attn_kernel<<<grd, blk, 0, stream>>>(Qbf, kvsTbf, V32, G, den, K32);
    }

    float* outF = (l == 0) ? G : out;
    ln_relu_kernel<<<TOT / 8, blk, 0, stream>>>(K32, ln_g[l], ln_b[l], outF, Hbf);
  }
}
